// Attention_10015863734737
// MI455X (gfx1250) — matrix operand order repaired and checked by reading, not compiled, not run on hardware
//
#include <hip/hip_runtime.h>

typedef __attribute__((ext_vector_type(16))) _Float16 v16h;
typedef __attribute__((ext_vector_type(8)))  float    v8f;

#define D_KEY    64
#define SEQLEN   4096
#define QBLK     128
#define KBLK     64
#define NWAVES   8
#define SCALE_INV 0.125f     // 1/sqrt(64)
#define NEG_BIG  1.0e9f

__device__ __forceinline__ v8f zero8() {
  v8f z = {0.f,0.f,0.f,0.f,0.f,0.f,0.f,0.f};
  return z;
}

// 16 contiguous f16 from LDS (B-matrix fragment): two b128 loads
__device__ __forceinline__ v16h ldB32(const _Float16* p) {
  union { uint4 u[2]; v16h h; } x;
  x.u[0] = *(const uint4*)(p);
  x.u[1] = *(const uint4*)(p + 8);
  return x.h;
}

// A-matrix fragment: halves[0..7] at p, halves[8..15] at p+8 (16 contiguous K)
__device__ __forceinline__ v16h ldA16(const _Float16* p) {
  union { uint4 u[2]; v16h h; } x;
  x.u[0] = *(const uint4*)(p);
  x.u[1] = *(const uint4*)(p + 8);
  return x.h;
}

__global__ __launch_bounds__(256, 2)
void fa_causal_wmma(const float* __restrict__ Q, const float* __restrict__ K,
                    const float* __restrict__ V, const float* __restrict__ Mk,
                    float* __restrict__ O)
{
  __shared__ _Float16 sK[KBLK][D_KEY];        // keys, row-major (f16)        8 KB
  __shared__ _Float16 sVt[D_KEY][KBLK];       // V transposed (f16)           8 KB
  __shared__ _Float16 sP[NWAVES][16][KBLK];   // per-wave P re-layout slab   16 KB
  __shared__ float    sM[KBLK];               // padding mask for key block

  const int tid  = threadIdx.x;
  const int lane = tid & 31;
  const int wave = tid >> 5;
  const int g    = lane >> 4;    // half-wave group (0/1)
  const int hn   = lane & 15;    // position within group

  const int nqb = SEQLEN / QBLK;
  const int b   = blockIdx.x / nqb;
  const int q0  = (blockIdx.x % nqb) * QBLK;
  const int qt0 = q0 + wave * 16;            // this wave's 16 query rows

  const float* Qb = Q  + ((size_t)b * SEQLEN) * D_KEY;
  const float* Kb = K  + ((size_t)b * SEQLEN) * D_KEY;
  const float* Vb = V  + ((size_t)b * SEQLEN) * D_KEY;
  const float* Mb = Mk + (size_t)b * SEQLEN;
  float*       Ob = O  + ((size_t)b * SEQLEN) * D_KEY;

  // ---- Q fragments in WMMA A-layout (f32 -> f16), resident all kernel ----
  // lane (g, hn): row M = hn; halves[0..7]: K = 32c + 16g + i; halves[8..15]: +8
  v16h qf[2];
  {
    const float* qrow = Qb + (size_t)(qt0 + hn) * D_KEY;
    #pragma unroll
    for (int c = 0; c < 2; ++c) {
      const float* p0 = qrow + 32 * c + 16 * g;
      const float* p1 = qrow + 32 * c + 8 + 16 * g;
      v16h t;
      #pragma unroll
      for (int i = 0; i < 8; ++i) {
        t[i]     = (_Float16)p0[i];
        t[i + 8] = (_Float16)p1[i];
      }
      qf[c] = t;
    }
  }

  v8f o[4] = { zero8(), zero8(), zero8(), zero8() };  // 16x64 f32 accum
  float mrow[8], lrow[8];
  #pragma unroll
  for (int r = 0; r < 8; ++r) { mrow[r] = -3.0e38f; lrow[r] = 0.f; }

  const int nkb  = (q0 + QBLK) / KBLK;   // causal bound for the 128-row block
  const int krow = tid >> 2;             // staging: key row 0..63
  const int kd0  = (tid & 3) * 16;       // staging: d offset 0..48

  for (int kb = 0; kb < nkb; ++kb) {
    const int key0 = kb * KBLK;

    // ---- cooperative staging: K row-major, V transposed, f32->f16 ----
    {
      const float* kp = Kb + (size_t)(key0 + krow) * D_KEY + kd0;
      const float* vp = Vb + (size_t)(key0 + krow) * D_KEY + kd0;
      #pragma unroll
      for (int c = 0; c < 4; ++c) {
        float4 kq = ((const float4*)kp)[c];
        float4 vq = ((const float4*)vp)[c];
        const int d = kd0 + 4 * c;
        sK[krow][d + 0] = (_Float16)kq.x;
        sK[krow][d + 1] = (_Float16)kq.y;
        sK[krow][d + 2] = (_Float16)kq.z;
        sK[krow][d + 3] = (_Float16)kq.w;
        sVt[d + 0][krow] = (_Float16)vq.x;
        sVt[d + 1][krow] = (_Float16)vq.y;
        sVt[d + 2][krow] = (_Float16)vq.z;
        sVt[d + 3][krow] = (_Float16)vq.w;
      }
      if (tid < KBLK) sM[tid] = Mb[key0 + tid];
      if (kb + 1 < nkb) {            // hint next block -> global_prefetch_b8
        __builtin_prefetch(kp + KBLK * D_KEY, 0, 0);
        __builtin_prefetch(vp + KBLK * D_KEY, 0, 0);
      }
    }
    __syncthreads();

    if (key0 <= qt0 + 15) {   // wave has at least one causally visible key
      // ---- S = Q K^T : four 16x16 tiles, K-dim 64 = 2 x wmma(K=32) each ----
      v8f s[4];
      #pragma unroll
      for (int t = 0; t < 4; ++t) {
        v16h kf0 = ldB32(&sK[16 * t + hn][16 * g]);        // d = 0..31
        v16h kf1 = ldB32(&sK[16 * t + hn][32 + 16 * g]);   // d = 32..63
        v8f acc = zero8();
        acc = __builtin_amdgcn_wmma_f32_16x16x32_f16(false, qf[0], false, kf0,
                                                     (short)0, acc, false, false);
        acc = __builtin_amdgcn_wmma_f32_16x16x32_f16(false, qf[1], false, kf1,
                                                     (short)0, acc, false, false);
        s[t] = acc;
      }

      // ---- masks folded into one FMA; causal only on diagonal blocks ----
      // logits = dot/8 - (1-mask)*1e9/8 [- 1e9/8 if col>row]
      float mpen[4];
      #pragma unroll
      for (int t = 0; t < 4; ++t)
        mpen[t] = (1.0f - sM[16 * t + hn]) * (NEG_BIG * SCALE_INV);

      const bool diag = (key0 + KBLK - 1) > qt0;
      if (diag) {
        #pragma unroll
        for (int t = 0; t < 4; ++t) {
          const int col = key0 + 16 * t + hn;
          #pragma unroll
          for (int r = 0; r < 8; ++r) {
            const int row = qt0 + r + 8 * g;
            float l = __builtin_fmaf(s[t][r], SCALE_INV, -mpen[t]);
            if (col > row) l -= NEG_BIG * SCALE_INV;
            s[t][r] = l;
          }
        }
      } else {
        #pragma unroll
        for (int t = 0; t < 4; ++t)
          #pragma unroll
          for (int r = 0; r < 8; ++r)
            s[t][r] = __builtin_fmaf(s[t][r], SCALE_INV, -mpen[t]);
      }

      // ---- online softmax (16-lane butterfly = one logit row) ----
      #pragma unroll
      for (int r = 0; r < 8; ++r) {
        float bm = fmaxf(fmaxf(s[0][r], s[1][r]), fmaxf(s[2][r], s[3][r]));
        #pragma unroll
        for (int x = 1; x < 16; x <<= 1) bm = fmaxf(bm, __shfl_xor(bm, x, 32));
        const float mnew  = fmaxf(mrow[r], bm);
        const float alpha = __expf(mrow[r] - mnew);
        float p[4];
        #pragma unroll
        for (int t = 0; t < 4; ++t) p[t] = __expf(s[t][r] - mnew);
        float rs = (p[0] + p[1]) + (p[2] + p[3]);
        #pragma unroll
        for (int x = 1; x < 16; x <<= 1) rs += __shfl_xor(rs, x, 32);
        lrow[r] = lrow[r] * alpha + rs;
        mrow[r] = mnew;
        #pragma unroll
        for (int to = 0; to < 4; ++to) o[to][r] *= alpha;
        // stage P (C/D layout -> LDS) for A-layout reload
        #pragma unroll
        for (int t = 0; t < 4; ++t)
          sP[wave][r + 8 * g][16 * t + hn] = (_Float16)p[t];
      }

      // ---- O += P(16x64) * V(64x64) : 8 wmmas ----
      v16h pa0 = ldA16(&sP[wave][hn][16 * g]);        // P cols 0..31
      v16h pa1 = ldA16(&sP[wave][hn][32 + 16 * g]);   // P cols 32..63
      #pragma unroll
      for (int to = 0; to < 4; ++to) {
        v16h vf0 = ldB32(&sVt[16 * to + hn][16 * g]);
        v16h vf1 = ldB32(&sVt[16 * to + hn][32 + 16 * g]);
        o[to] = __builtin_amdgcn_wmma_f32_16x16x32_f16(false, pa0, false, vf0,
                                                       (short)0, o[to], false, false);
        o[to] = __builtin_amdgcn_wmma_f32_16x16x32_f16(false, pa1, false, vf1,
                                                       (short)0, o[to], false, false);
      }
    }
    __syncthreads();   // protect sK/sVt before next staging round
  }

  // ---- finalize: 1/l, dead-row zeroing, store ----
  const float athresh = -(NEG_BIG * 0.5f) * SCALE_INV;  // "any dot > -NEG_BIG/2"
  #pragma unroll
  for (int r = 0; r < 8; ++r) {
    const float alive = (mrow[r] > athresh) ? 1.0f : 0.0f;
    const float inv   = (lrow[r] > 0.f) ? (alive / lrow[r]) : 0.f;
    float* orow = Ob + (size_t)(qt0 + r + 8 * g) * D_KEY + hn;
    #pragma unroll
    for (int to = 0; to < 4; ++to)
      orow[16 * to] = o[to][r] * inv;
  }
}

extern "C" void kernel_launch(void* const* d_in, const int* in_sizes, int n_in,
                              void* d_out, int out_size, void* d_ws, size_t ws_size,
                              hipStream_t stream) {
  const float* Q  = (const float*)d_in[0];
  const float* K  = (const float*)d_in[1];
  const float* V  = (const float*)d_in[2];
  const float* Mk = (const float*)d_in[3];
  float* O = (float*)d_out;
  const int B = in_sizes[0] / (SEQLEN * D_KEY);
  dim3 grid(B * (SEQLEN / QBLK));
  fa_causal_wmma<<<grid, 256, 0, stream>>>(Q, K, V, Mk, O);
}
